// ChangeDetectionMamba_39633958208120
// MI455X (gfx1250) — compile-verified, matrix operand order checked by reading, not run on hardware
//
#include <hip/hip_runtime.h>
#include <math.h>

#define BATCH   8
#define SEQL    4096
#define DIMC    552
#define EXPE    1104
#define DSS     16
#define DCONV   4
#define TOPK    409

#define KPAD1   576            // 552 padded to 32*18
#define KSTEPS1 18
#define NT1     69             // 1104 / 16
#define NB1     9              // ceil(69/8) n-blocks
#define M1      (BATCH*SEQL)   // 32768

#define M2      (BATCH*TOPK)   // 3272
#define M2P     3280           // padded to 16*205
#define K2PAD   1120           // 1104 padded to 32*35
#define KSTEPS2 35
#define NT2     35             // ceil(552/16) -> 560
#define NB2     5              // ceil(35/8)

typedef __attribute__((ext_vector_type(16))) __bf16 v16bf;
typedef __attribute__((ext_vector_type(8)))  __bf16 v8bf;
typedef __attribute__((ext_vector_type(8)))  float  v8f;
typedef __attribute__((ext_vector_type(4)))  int    v4i;

#if defined(__gfx1250__) && __has_builtin(__builtin_amdgcn_global_load_async_to_lds_b128)
#define HAVE_ASYNC_LDS 1
#endif

__device__ __forceinline__ void wait_asynccnt0() {
#if defined(__gfx1250__)
#if __has_builtin(__builtin_amdgcn_s_wait_asynccnt)
  __builtin_amdgcn_s_wait_asynccnt(0);
#else
  asm volatile("s_wait_asynccnt 0x0" ::: "memory");
#endif
#endif
}

#ifdef HAVE_ASYNC_LDS
// async copy 16 bytes global -> LDS (tracked by ASYNCcnt)
__device__ __forceinline__ void async_copy16(const __bf16* g, __bf16* l) {
  __builtin_amdgcn_global_load_async_to_lds_b128(
      (__attribute__((address_space(1))) v4i*)const_cast<__bf16*>(g),
      (__attribute__((address_space(3))) v4i*)l, 0, 0);
}
#endif

// ---------------------------------------------------------------- RMS norm
// one block per row; writes bf16 row padded to KPAD1 with zeros
__global__ void k_rmsnorm(const float* __restrict__ x, const float* __restrict__ nw,
                          void* __restrict__ xn_) {
  __bf16* xn = (__bf16*)xn_;
  const int row = blockIdx.x;
  const int tid = threadIdx.x;
  __shared__ float red[256];
  const float* xr = x + (size_t)row * DIMC;
  const int c0 = tid, c1 = tid + 256, c2 = tid + 512;
  float v0 = (c0 < DIMC) ? xr[c0] : 0.f;
  float v1 = (c1 < DIMC) ? xr[c1] : 0.f;
  float v2 = (c2 < DIMC) ? xr[c2] : 0.f;
  red[tid] = v0*v0 + v1*v1 + v2*v2;
  __syncthreads();
  for (int s = 128; s > 0; s >>= 1) {
    if (tid < s) red[tid] += red[tid + s];
    __syncthreads();
  }
  const float rms = sqrtf(red[0]) * rsqrtf((float)DIMC);
  const float inv = 1.0f / (rms + 1e-6f);
  __bf16* xo = xn + (size_t)row * KPAD1;
  if (c0 < KPAD1) xo[c0] = (c0 < DIMC) ? (__bf16)(v0 * nw[c0] * inv) : (__bf16)0.0f;
  if (c1 < KPAD1) xo[c1] = (c1 < DIMC) ? (__bf16)(v1 * nw[c1] * inv) : (__bf16)0.0f;
  if (c2 < KPAD1) xo[c2] = (c2 < DIMC) ? (__bf16)(v2 * nw[c2] * inv) : (__bf16)0.0f;
}

// ---------------------------------------------------------------- B-fragment packer
// packs w (N x K row-major, f32) as bf16 WMMA B fragments: B[k][n] = w[n][k].
// fragment (nt, ks) = 1024 bf16; lane l stores 16 contiguous elems at l*16.
// lane l<16: n=l,  K in {0..7, 16..23};  lane l>=16: n=l-16, K in {8..15, 24..31}
__global__ void k_pack_b(const float* __restrict__ w, void* __restrict__ frag_,
                         int N, int K, int ksteps) {
  __bf16* frag = (__bf16*)frag_;
  const int nt = blockIdx.x / ksteps;
  const int ks = blockIdx.x % ksteps;
  __bf16* fb = frag + (size_t)blockIdx.x * 1024;
  for (int f = threadIdx.x; f < 1024; f += 256) {
    const int l = f >> 4, j = f & 15;
    const int n = nt * 16 + (l & 15);
    const int k = ks * 32 + j + ((j & 8) ? 8 : 0) + ((l & 16) ? 8 : 0);
    const float v = (n < N && k < K) ? w[(size_t)n * K + k] : 0.0f;
    fb[f] = (__bf16)v;
  }
}

// ---------------------------------------------------------------- WMMA GEMM
// C[m,n] = sum_k A[m,k]*B[k,n] + bias[n].  A: bf16, row-stride KPADN (zero padded).
// grid = (Mtiles, ceil(NTN/8)); one block = 8 waves; each wave owns ONE 16x16 tile,
// so no loop-invariant fragment hoisting / scratch spills in the k-loop.
template <int KSTEPSN, int KPADN, int NTN, int NOUT>
__global__ void k_gemm(const void* __restrict__ A_, const void* __restrict__ Bfrag_,
                       const float* __restrict__ bias, float* __restrict__ C,
                       int Mvalid) {
  const __bf16* A     = (const __bf16*)A_;
  const __bf16* Bfrag = (const __bf16*)Bfrag_;
  __shared__ __align__(16) __bf16 As[16 * KPADN];

  const int m0  = blockIdx.x * 16;
  const int tid = threadIdx.x;
  const int cpr = KPADN / 8;           // 16-byte chunks per row

  // stage the 16-row A tile into LDS
#ifdef HAVE_ASYNC_LDS
  for (int c = tid; c < 16 * cpr; c += 256) {
    const int r   = c / cpr;
    const int col = (c % cpr) * 8;
    async_copy16(&A[(size_t)(m0 + r) * KPADN + col], &As[r * KPADN + col]);
  }
  wait_asynccnt0();
#else
  for (int c = tid; c < 16 * cpr; c += 256) {
    const int r   = c / cpr;
    const int col = (c % cpr) * 8;
    *(v8bf*)&As[r * KPADN + col] = *(const v8bf*)&A[(size_t)(m0 + r) * KPADN + col];
  }
#endif
  __syncthreads();

  const int wave = tid >> 5, lane = tid & 31;
  const int nt   = blockIdx.y * 8 + wave;          // wave-uniform
  if (nt < NTN) {
    const int rowA = lane & 15;
    const int koff = (lane & 16) ? 8 : 0;
    v8f acc = {};
    const __bf16* bf = Bfrag + (size_t)nt * KSTEPSN * 1024 + (size_t)lane * 16;
#pragma unroll
    for (int ks = 0; ks < KSTEPSN; ++ks) {
      union { v16bf v; v8bf h[2]; } a;
      const int kb = ks * 32 + koff;
      a.h[0] = *(const v8bf*)&As[rowA * KPADN + kb];
      a.h[1] = *(const v8bf*)&As[rowA * KPADN + kb + 16];
      const v16bf b = *(const v16bf*)(bf + (size_t)ks * 1024);
      if (ks + 1 < KSTEPSN) __builtin_prefetch(bf + (size_t)(ks + 1) * 1024, 0, 1);
      acc = __builtin_amdgcn_wmma_f32_16x16x32_bf16(false, a.v, false, b,
                                                    (short)0, acc, false, false);
    }
    const int n = nt * 16 + (lane & 15);
    if (n < NOUT) {
      const float bs = bias[n];
      const int mbase = m0 + ((lane >> 4) << 3);
#pragma unroll
      for (int r = 0; r < 8; ++r) {
        const int m = mbase + r;
        if (m < Mvalid) C[(size_t)m * NOUT + n] = acc[r] + bs;
      }
    }
  }
}

// ---------------------------------------------------------------- cosine sim vs center
__global__ void k_sim(const float* __restrict__ xp, float* __restrict__ sim) {
  const int b = blockIdx.x >> 12;
  const int l = blockIdx.x & (SEQL - 1);
  const float* xr = xp + ((size_t)b * SEQL + l) * EXPE;
  const float* cr = xp + ((size_t)b * SEQL + SEQL / 2) * EXPE;
  __shared__ float rd[3][256];
  float d = 0.f, nx = 0.f, nc = 0.f;
  for (int e = threadIdx.x; e < EXPE; e += 256) {
    const float a = xr[e], c = cr[e];
    d += a * c; nx += a * a; nc += c * c;
  }
  rd[0][threadIdx.x] = d; rd[1][threadIdx.x] = nx; rd[2][threadIdx.x] = nc;
  __syncthreads();
  for (int s = 128; s > 0; s >>= 1) {
    if (threadIdx.x < s) {
      rd[0][threadIdx.x] += rd[0][threadIdx.x + s];
      rd[1][threadIdx.x] += rd[1][threadIdx.x + s];
      rd[2][threadIdx.x] += rd[2][threadIdx.x + s];
    }
    __syncthreads();
  }
  if (threadIdx.x == 0) {
    const float den = fmaxf(sqrtf(rd[1][0]), 1e-12f) * fmaxf(sqrtf(rd[2][0]), 1e-12f);
    sim[blockIdx.x] = rd[0][0] / den;
  }
}

// ---------------------------------------------------------------- top-k via bitonic sort
// softmax is monotone -> top-k on raw cosine sim == top-k on softmax(sim).
// descending, ties broken by lower index (matches jax.lax.top_k).
__global__ void k_topk(const float* __restrict__ sim, int* __restrict__ idxout) {
  __shared__ float key[SEQL];
  __shared__ int   val[SEQL];
  const int b = blockIdx.x, tid = threadIdx.x;
  for (int i = tid; i < SEQL; i += 256) { key[i] = sim[b * SEQL + i]; val[i] = i; }
  for (int size = 2; size <= SEQL; size <<= 1) {
    for (int stride = size >> 1; stride > 0; stride >>= 1) {
      __syncthreads();
      for (int i = tid; i < SEQL / 2; i += 256) {
        const int pos = (i << 1) - (i & (stride - 1));
        const int par = pos + stride;
        const bool desc = ((pos & size) == 0);
        const float k0 = key[pos], k1 = key[par];
        const int v0 = val[pos], v1 = val[par];
        // "less" = smaller key, tie -> larger index
        const bool less01 = (k0 < k1) || (k0 == k1 && v0 > v1);
        const bool doswap = desc ? less01 : !less01;
        if (doswap) { key[pos] = k1; key[par] = k0; val[pos] = v1; val[par] = v0; }
      }
    }
  }
  __syncthreads();
  for (int t = tid; t < TOPK; t += 256) idxout[b * TOPK + t] = val[t];
}

// ---------------------------------------------------------------- gather + causal depthwise conv
__global__ void k_conv(const float* __restrict__ xp, const int* __restrict__ idx,
                       const float* __restrict__ cw, float* __restrict__ xconv) {
  const int blk = blockIdx.x;            // b*TOPK + t
  const int b = blk / TOPK, t = blk % TOPK;
  const int* idb = idx + b * TOPK;
  const float* rows[DCONV];
#pragma unroll
  for (int j = 0; j < DCONV; ++j) {
    const int tt = t - (DCONV - 1) + j;
    rows[j] = (tt >= 0) ? xp + ((size_t)b * SEQL + idb[tt]) * EXPE : nullptr;
  }
  float* out = xconv + (size_t)blk * EXPE;
  for (int e = threadIdx.x; e < EXPE; e += 256) {
    float acc = 0.f;
#pragma unroll
    for (int j = 0; j < DCONV; ++j)
      if (rows[j]) acc += cw[e * DCONV + j] * rows[j][e];
    out[e] = acc;
  }
}

// ---------------------------------------------------------------- zero fill (dwords)
__global__ void k_zero(unsigned int* __restrict__ p, int n) {
  const int i = blockIdx.x * 256 + threadIdx.x;
  if (i < n) p[i] = 0u;
}

// ---------------------------------------------------------------- SSM scan
// one thread per (b, e); h[16] in registers; writes bf16 into padded GEMM2-A buffer
__global__ void k_scan(const float* __restrict__ xconv, const float* __restrict__ A,
                       const float* __restrict__ Bp, const float* __restrict__ Cp,
                       void* __restrict__ xscan_) {
  __bf16* xscan = (__bf16*)xscan_;
  __shared__ float As[DSS * DSS];
  if (threadIdx.x < DSS * DSS) As[threadIdx.x] = A[threadIdx.x];
  __syncthreads();
  const int gi = blockIdx.x * 256 + threadIdx.x;
  if (gi >= BATCH * EXPE) return;
  const int b = gi / EXPE, e = gi % EXPE;
  float sB[DSS], sC[DSS], h[DSS];
#pragma unroll
  for (int s = 0; s < DSS; ++s) {
    sB[s] = 1.f / (1.f + expf(-Bp[s]));
    sC[s] = 1.f / (1.f + expf(-Cp[e * DSS + s]));
    h[s] = 0.f;
  }
  for (int t = 0; t < TOPK; ++t) {
    const float xt = xconv[((size_t)b * TOPK + t) * EXPE + e];
    float hn[DSS];
#pragma unroll
    for (int i = 0; i < DSS; ++i) {
      float a = sB[i] * xt;
#pragma unroll
      for (int j = 0; j < DSS; ++j) a += h[j] * As[i * DSS + j];
      hn[i] = a;
    }
    float y = 0.f;
#pragma unroll
    for (int i = 0; i < DSS; ++i) { h[i] = hn[i]; y += hn[i] * sC[i]; }
    xscan[((size_t)b * TOPK + t) * K2PAD + e] = (__bf16)y;
  }
}

// ---------------------------------------------------------------- residual copy + scatter
__global__ void k_copy(const float* __restrict__ x, float* __restrict__ out, int n4) {
  const int i = blockIdx.x * 256 + threadIdx.x;
  if (i < n4) ((float4*)out)[i] = ((const float4*)x)[i];
}

__global__ void k_scatter(const float* __restrict__ xproc, const int* __restrict__ idx,
                          float* __restrict__ out) {
  const int blk = blockIdx.x;
  const int b = blk / TOPK, t = blk % TOPK;
  const int l = idx[b * TOPK + t];
  float* o = out + ((size_t)b * SEQL + l) * DIMC;
  const float* p = xproc + (size_t)blk * DIMC;
  for (int c = threadIdx.x; c < DIMC; c += 256) o[c] += p[c];
}

// ---------------------------------------------------------------- launch
extern "C" void kernel_launch(void* const* d_in, const int* in_sizes, int n_in,
                              void* d_out, int out_size, void* d_ws, size_t ws_size,
                              hipStream_t stream) {
  const float* x     = (const float*)d_in[0];
  const float* nw    = (const float*)d_in[1];
  const float* w_in  = (const float*)d_in[2];
  const float* b_in  = (const float*)d_in[3];
  const float* w_out = (const float*)d_in[4];
  const float* b_out = (const float*)d_in[5];
  const float* A     = (const float*)d_in[6];
  const float* Bp    = (const float*)d_in[7];
  const float* Cp    = (const float*)d_in[8];
  const float* cw    = (const float*)d_in[9];
  float* out = (float*)d_out;

  char* ws = (char*)d_ws;
  size_t off = 0;
  auto carve = [&](size_t bytes) -> void* {
    void* p = ws + off;
    off = (off + bytes + 255) & ~(size_t)255;
    return p;
  };
  void*  xn       = carve((size_t)M1 * KPAD1 * 2);        // bf16 normed x, padded K
  void*  winfrag  = carve((size_t)NT1 * KSTEPS1 * 2048);  // bf16 fragments
  float* xproj    = (float*)carve((size_t)M1 * EXPE * 4);
  float* sim      = (float*)carve((size_t)M1 * 4);
  int*   idx      = (int*)carve((size_t)BATCH * TOPK * 4);
  float* xconv    = (float*)carve((size_t)M2 * EXPE * 4);
  void*  xscan    = carve((size_t)M2P * K2PAD * 2);       // bf16, zero-padded
  void*  woutfrag = carve((size_t)NT2 * KSTEPS2 * 2048);
  float* xproc    = (float*)carve((size_t)M2 * DIMC * 4);
  (void)ws_size; (void)in_sizes; (void)n_in; (void)out_size;

  // 1. RMS norm -> bf16
  k_rmsnorm<<<M1, 256, 0, stream>>>(x, nw, xn);
  // 2. pack w_in fragments
  k_pack_b<<<NT1 * KSTEPS1, 256, 0, stream>>>(w_in, winfrag, EXPE, DIMC, KSTEPS1);
  // 3. x_proj GEMM (WMMA bf16), 2D grid: one 16x16 tile per wave
  k_gemm<KSTEPS1, KPAD1, NT1, EXPE>
      <<<dim3(M1 / 16, NB1), 256, 0, stream>>>(xn, winfrag, b_in, xproj, M1);
  // 4. cosine similarity vs center token
  k_sim<<<M1, 256, 0, stream>>>(xproj, sim);
  // 5. exact top-k (bitonic sort per batch)
  k_topk<<<BATCH, 256, 0, stream>>>(sim, idx);
  // 6. gather + causal depthwise conv
  k_conv<<<M2, 256, 0, stream>>>(xproj, idx, cw, xconv);
  // 7. zero padded scan output buffer, then SSM scan
  {
    const int ndw = (M2P * K2PAD) / 2;   // bf16 pairs as dwords
    k_zero<<<(ndw + 255) / 256, 256, 0, stream>>>((unsigned int*)xscan, ndw);
  }
  k_scan<<<(BATCH * EXPE + 255) / 256, 256, 0, stream>>>(xconv, A, Bp, Cp, xscan);
  // 8. pack w_out fragments + output GEMM (WMMA bf16)
  k_pack_b<<<NT2 * KSTEPS2, 256, 0, stream>>>(w_out, woutfrag, DIMC, EXPE, KSTEPS2);
  k_gemm<KSTEPS2, K2PAD, NT2, DIMC>
      <<<dim3(M2P / 16, NB2), 256, 0, stream>>>(xscan, woutfrag, b_out, xproc, M2);
  // 9. residual copy + scatter
  {
    const int n4 = (BATCH * SEQL * DIMC) / 4;
    k_copy<<<(n4 + 255) / 256, 256, 0, stream>>>(x, out, n4);
  }
  k_scatter<<<M2, 256, 0, stream>>>(xproc, idx, out);
}
